// GNNEncoder_64115271794802
// MI455X (gfx1250) — matrix operand order repaired and checked by reading, not compiled, not run on hardware
//
#include <hip/hip_runtime.h>
#include <math.h>

// ---------------- problem constants (from reference) ----------------
#define N_NODES   50000
#define HEADS     4
#define HID       32
#define D1        128          // HEADS*HID
#define NEG_SLOPE 0.2f
#define KTILE     64           // K-tile DMA'd per TDM descriptor pair

typedef __attribute__((ext_vector_type(16))) _Float16     v16h;
typedef __attribute__((ext_vector_type(8)))  float        v8f;
typedef __attribute__((ext_vector_type(4)))  unsigned int u32x4;
typedef __attribute__((ext_vector_type(8)))  int          i32x8;
typedef __attribute__((ext_vector_type(4)))  int          i32x4;

union AFrag { v16h v; _Float16 e[16]; };
union CFrag { v8f  v; float    e[8];  };

// flat (generic) pointer -> 32-bit LDS byte offset (addrspacecast to AS(3),
// which the backend lowers to taking the low 32 bits / LDS offset)
#define LDS_AS __attribute__((address_space(3)))
__device__ __forceinline__ unsigned lds_addr_of(const void* p) {
  return (unsigned)(unsigned long long)(LDS_AS const void*)p;
}

// ---------------------------------------------------------------
// Issue one 2-D Tensor-Data-Mover load: tile1 rows x tile0 elems of
// 16-bit data, row stride = stride0 elements, into contiguous LDS.
// D# bit layout per CDNA5 ISA ch.10.8 (group0: count/lds/global/type,
// group1: data_size + dims/tile/stride).  Tracked by TENSORcnt.
// ---------------------------------------------------------------
__device__ __forceinline__ void tdm_load_2d_f16(unsigned lds_off,
                                                const void* gptr,
                                                unsigned dim0, unsigned dim1,
                                                unsigned tile0, unsigned tile1,
                                                unsigned stride0) {
  unsigned long long ga = (unsigned long long)gptr;
  u32x4 g0;
  g0.x = 1u;                                            // count=1 (valid user D#)
  g0.y = lds_off;                                       // lds_addr [63:32]
  g0.z = (unsigned)(ga & 0xFFFFFFFFu);                  // global_addr low
  g0.w = (unsigned)((ga >> 32) & 0x1FFFFFFu)            // global_addr[56:32]
         | (2u << 30);                                  // type=2 ("image")
  i32x8 g1;
  g1.s0 = 0x00010000;                                   // data_size=1 -> 2 bytes
  g1.s1 = (int)(dim0 << 16);                            // tensor_dim0[15:0] @ [63:48]
  g1.s2 = (int)((dim0 >> 16) | (dim1 << 16));           // dim0 hi | dim1 lo
  g1.s3 = (int)((dim1 >> 16) | (tile0 << 16));          // dim1 hi | tile_dim0
  g1.s4 = (int)(tile1 & 0xFFFFu);                       // tile_dim1 (tile_dim2=0)
  g1.s5 = (int)stride0;                                 // tensor_dim0_stride lo32
  g1.s6 = 0;                                            // stride0 hi | stride1 lo
  g1.s7 = 0;
  i32x4 z4 = {0, 0, 0, 0};                              // groups 2/3: unused (2-D)
#if defined(__clang_major__) && (__clang_major__ >= 23)
  i32x8 z8 = {0, 0, 0, 0, 0, 0, 0, 0};
  __builtin_amdgcn_tensor_load_to_lds(g0, g1, z4, z4, z8, 0);
#else
  __builtin_amdgcn_tensor_load_to_lds(g0, g1, z4, z4, 0);
#endif
}

// ---------------------------------------------------------------
// fp32 -> fp16 elementwise convert
// ---------------------------------------------------------------
__global__ void cvt_f16_kernel(const float* __restrict__ in,
                               _Float16* __restrict__ out, int n) {
  int t = blockIdx.x * blockDim.x + threadIdx.x;
  if (t < n) out[t] = (_Float16)in[t];
}

// W [K x 128] fp32 row-major -> WT [128 x K] fp16 (transposed),
// so B-matrix fragments are K-contiguous in LDS after the TDM copy.
__global__ void transpose_w_kernel(const float* __restrict__ W,
                                   _Float16* __restrict__ WT, int K) {
  int t = blockIdx.x * blockDim.x + threadIdx.x;
  if (t >= K * D1) return;
  int k = t >> 7;
  int m = t & 127;
  WT[(size_t)m * K + k] = (_Float16)W[t];
}

// ---------------------------------------------------------------
// GEMM: C[n x 128] = A16[n x K] * W (given as WT16[128 x K]), f16 WMMA.
// Block = 256 threads = 8 wave32; 16-row x 128-col stripe per block.
// Per KTILE: wave0 DMAs A-tile (16 x 64) and W-tile (128 x 64) into LDS
// via the Tensor Data Mover, waits TENSORcnt, workgroup barrier, then
// each wave runs 2 x v_wmma_f32_16x16x32_f16 on its 16x16 tile.
// ---------------------------------------------------------------
__global__ __launch_bounds__(256)
void gemm_wmma_tdm(const _Float16* __restrict__ A16,
                   const _Float16* __restrict__ WT16,
                   float* __restrict__ C,
                   int K) {
  __shared__ __align__(128) _Float16 As[16][KTILE];     //  2 KB
  __shared__ __align__(128) _Float16 WsT[128][KTILE];   // 16 KB

  const int t    = threadIdx.x;
  const int wave = t >> 5;             // 0..7  (wave32)
  const int lane = t & 31;
  const int half = lane >> 4;
  const int l16  = lane & 15;
  const int row0 = blockIdx.x * 16;

  const unsigned as_off = lds_addr_of(&As[0][0]);
  const unsigned ws_off = lds_addr_of(&WsT[0][0]);

  CFrag c;
#pragma unroll
  for (int i = 0; i < 8; ++i) c.e[i] = 0.0f;

  for (int k0 = 0; k0 < K; k0 += KTILE) {
    __syncthreads();                   // protect LDS from previous iteration
    if (wave == 0) {                   // uniform branch: one wave drives the TDM
      // A tile: 16 rows x KTILE, row stride K
      tdm_load_2d_f16(as_off, A16 + (size_t)row0 * K + k0,
                      (unsigned)K, 16u, KTILE, 16u, (unsigned)K);
      // W tile: 128 rows (output cols) x KTILE, row stride K
      tdm_load_2d_f16(ws_off, WT16 + k0,
                      (unsigned)K, 128u, KTILE, 128u, (unsigned)K);
      __builtin_amdgcn_s_wait_tensorcnt(0);
    }
    __syncthreads();

#pragma unroll
    for (int ks = 0; ks < KTILE; ks += 32) {
      // A fragment (ISA 16-bit A 16x32 layout):
      // lanes 0-15: K = {0..7, 16..23}; lanes 16-31: K = {8..15, 24..31}
      AFrag a, b;
#pragma unroll
      for (int i = 0; i < 8; ++i)  a.e[i]     = As[l16][ks + half * 8 + i];
#pragma unroll
      for (int i = 0; i < 8; ++i)  a.e[8 + i] = As[l16][ks + 16 + half * 8 + i];
      // B fragment (32x16): lanes 0-15 hold K {0..7, 16..23}, lanes 16-31 K {8..15, 24..31}
      const int col = wave * 16 + l16;
#pragma unroll
      for (int i = 0; i < 8; ++i)  b.e[i]     = WsT[col][ks + half * 8 + i];
#pragma unroll
      for (int i = 0; i < 8; ++i)  b.e[8 + i] = WsT[col][ks + 16 + half * 8 + i];

      c.v = __builtin_amdgcn_wmma_f32_16x16x32_f16(
          false, a.v, false, b.v, (short)0, c.v, false, false);
    }
  }

  // C/D layout: VGPR r -> row (r + 8*half), col = lane&15 within tile
  const int col = wave * 16 + l16;
#pragma unroll
  for (int r = 0; r < 8; ++r)
    C[(size_t)(row0 + half * 8 + r) * D1 + col] = c.e[r];
}

// ---------------------------------------------------------------
// alpha_src / alpha_dst : per (node, head) dot of h with attention vecs
// ---------------------------------------------------------------
__global__ void alpha_kernel(const float* __restrict__ h,
                             const float* __restrict__ a_src,
                             const float* __restrict__ a_dst,
                             float* __restrict__ as,
                             float* __restrict__ ad) {
  int t = blockIdx.x * blockDim.x + threadIdx.x;
  if (t >= N_NODES * HEADS) return;
  int n  = t >> 2;
  int hh = t & 3;
  const float* hp = h + (size_t)n * D1 + hh * HID;
  float s = 0.f, d = 0.f;
#pragma unroll
  for (int c = 0; c < HID; ++c) {
    float v = hp[c];
    s += v * a_src[hh * HID + c];
    d += v * a_dst[hh * HID + c];
  }
  as[t] = s;
  ad[t] = d;
}

__global__ void fill_kernel(float* __restrict__ p, float val, int n) {
  int t = blockIdx.x * blockDim.x + threadIdx.x;
  if (t < n) p[t] = val;
}

__device__ __forceinline__ void atomicMaxFloat(float* addr, float v) {
  if (v >= 0.f)
    atomicMax((int*)addr, __float_as_int(v));
  else
    atomicMin((unsigned int*)addr, __float_as_uint(v));
}

__device__ __forceinline__ void edge_endpoints(const int* __restrict__ ei,
                                               int E_in, int e,
                                               int& src, int& dst) {
  if (e < E_in) { src = ei[e]; dst = ei[E_in + e]; }
  else          { src = dst = e - E_in; }          // appended self loops
}

__device__ __forceinline__ float leaky(float x) {
  return x > 0.f ? x : NEG_SLOPE * x;
}

// pass 1: segment max of leaky-relu logits over destination
__global__ void edge_max_kernel(const int* __restrict__ ei, int E_in,
                                const float* __restrict__ as,
                                const float* __restrict__ ad,
                                float* __restrict__ segmax) {
  int t = blockIdx.x * blockDim.x + threadIdx.x;
  int E = E_in + N_NODES;
  if (t >= E * HEADS) return;
  int e = t >> 2, hh = t & 3;
  int src, dst;
  edge_endpoints(ei, E_in, e, src, dst);
  float lg = leaky(as[src * HEADS + hh] + ad[dst * HEADS + hh]);
  atomicMaxFloat(&segmax[dst * HEADS + hh], lg);
}

// seg_max = where(isfinite(seg_max), seg_max, 0)
__global__ void clean_max_kernel(float* __restrict__ segmax) {
  int t = blockIdx.x * blockDim.x + threadIdx.x;
  if (t >= N_NODES * HEADS) return;
  float m = segmax[t];
  if (!__builtin_isfinite(m)) segmax[t] = 0.f;
}

// pass 2: denom = segment_sum(exp(logit - segmax[dst]))
__global__ void edge_denom_kernel(const int* __restrict__ ei, int E_in,
                                  const float* __restrict__ as,
                                  const float* __restrict__ ad,
                                  const float* __restrict__ segmax,
                                  float* __restrict__ denom) {
  int t = blockIdx.x * blockDim.x + threadIdx.x;
  int E = E_in + N_NODES;
  if (t >= E * HEADS) return;
  int e = t >> 2, hh = t & 3;
  int src, dst;
  edge_endpoints(ei, E_in, e, src, dst);
  float lg = leaky(as[src * HEADS + hh] + ad[dst * HEADS + hh]);
  float w  = __expf(lg - segmax[dst * HEADS + hh]);
  atomicAdd(&denom[dst * HEADS + hh], w);
}

// pass 3: acc[dst, :] += h[src, :] * attn  — one thread per (edge, channel)
// 128 consecutive lanes cover one edge -> coalesced gather/scatter,
// accumulator stays L2-resident so atomics run at the L2 atomic units
__global__ void edge_agg_kernel(const int* __restrict__ ei, int E_in,
                                const float* __restrict__ as,
                                const float* __restrict__ ad,
                                const float* __restrict__ segmax,
                                const float* __restrict__ denom,
                                const float* __restrict__ h,
                                float* __restrict__ acc) {
  long long t = (long long)blockIdx.x * blockDim.x + threadIdx.x;
  long long total = (long long)(E_in + N_NODES) * D1;
  if (t >= total) return;
  int e  = (int)(t >> 7);
  int j  = (int)(t & 127);     // channel within node feature = hh*32 + c
  int hh = j >> 5;
  int src, dst;
  edge_endpoints(ei, E_in, e, src, dst);
  float lg   = leaky(as[src * HEADS + hh] + ad[dst * HEADS + hh]);
  float w    = __expf(lg - segmax[dst * HEADS + hh]);
  float attn = w / (denom[dst * HEADS + hh] + 1e-16f);
  float val  = h[(size_t)src * D1 + j] * attn;
  atomicAdd(&acc[(size_t)dst * D1 + j], val);
}

// bias add (+ optional ELU, in place safe)
__global__ void bias_elu_kernel(const float* __restrict__ acc,
                                const float* __restrict__ bias,
                                float* __restrict__ out, int apply_elu) {
  int t = blockIdx.x * blockDim.x + threadIdx.x;
  if (t >= N_NODES * D1) return;
  float v = acc[t] + bias[t & 127];
  if (apply_elu) v = v > 0.f ? v : expm1f(v);
  out[t] = v;
}

static inline int cdiv(long long a, int b) { return (int)((a + b - 1) / b); }

extern "C" void kernel_launch(void* const* d_in, const int* in_sizes, int n_in,
                              void* d_out, int out_size, void* d_ws, size_t ws_size,
                              hipStream_t stream) {
  const float* x      = (const float*)d_in[0];
  const int*   ei     = (const int*)  d_in[1];
  const float* W1     = (const float*)d_in[2];
  const float* a_src1 = (const float*)d_in[3];
  const float* a_dst1 = (const float*)d_in[4];
  const float* b1     = (const float*)d_in[5];
  const float* W2     = (const float*)d_in[6];
  const float* a_src2 = (const float*)d_in[7];
  const float* a_dst2 = (const float*)d_in[8];
  const float* b2     = (const float*)d_in[9];
  float* out = (float*)d_out;

  const int E_in = in_sizes[1] / 2;          // 800000
  const int E    = E_in + N_NODES;           // + self loops
  const int NH   = N_NODES * HEADS;
  const int ND   = N_NODES * D1;
  const int NK1  = N_NODES * 256;

  // workspace layout
  float* ws    = (float*)d_ws;
  float* h_buf = ws;                 // [N,128]  h1, later h2
  float* x2    = h_buf + ND;         // [N,128]  layer-1 aggregate -> layer-2 input
  float* asb   = x2 + ND;            // [N,4]
  float* adb   = asb + NH;           // [N,4]
  float* smax  = adb + NH;           // [N,4]
  float* den   = smax + NH;          // [N,4]
  _Float16* xf16 = (_Float16*)(den + NH);    // [N,256] f16 (reused as [N,128])
  _Float16* wT16 = xf16 + NK1;               // [128,256] f16 (reused [128,128])
  (void)ws_size; (void)n_in;

  const int TB = 256;
  dim3 blk(TB);
  const int gemm_blocks = N_NODES / 16;                 // 3125
  const long long eagg  = (long long)E * D1;

  // ---------------- layer 1 ----------------
  cvt_f16_kernel   <<<cdiv(NK1, TB), blk, 0, stream>>>(x, xf16, NK1);
  transpose_w_kernel<<<cdiv(256 * D1, TB), blk, 0, stream>>>(W1, wT16, 256);
  gemm_wmma_tdm    <<<gemm_blocks, blk, 0, stream>>>(xf16, wT16, h_buf, 256);

  alpha_kernel<<<cdiv(NH, TB), blk, 0, stream>>>(h_buf, a_src1, a_dst1, asb, adb);

  fill_kernel<<<cdiv(NH, TB), blk, 0, stream>>>(smax, -INFINITY, NH);
  fill_kernel<<<cdiv(NH, TB), blk, 0, stream>>>(den, 0.0f, NH);
  fill_kernel<<<cdiv(ND, TB), blk, 0, stream>>>(x2, 0.0f, ND);

  edge_max_kernel  <<<cdiv((long long)E * HEADS, TB), blk, 0, stream>>>(ei, E_in, asb, adb, smax);
  clean_max_kernel <<<cdiv(NH, TB), blk, 0, stream>>>(smax);
  edge_denom_kernel<<<cdiv((long long)E * HEADS, TB), blk, 0, stream>>>(ei, E_in, asb, adb, smax, den);
  edge_agg_kernel  <<<cdiv(eagg, TB), blk, 0, stream>>>(ei, E_in, asb, adb, smax, den, h_buf, x2);

  bias_elu_kernel<<<cdiv(ND, TB), blk, 0, stream>>>(x2, b1, x2, /*elu=*/1);

  // ---------------- layer 2 ----------------
  cvt_f16_kernel   <<<cdiv(ND, TB), blk, 0, stream>>>(x2, xf16, ND);
  transpose_w_kernel<<<cdiv(128 * D1, TB), blk, 0, stream>>>(W2, wT16, 128);
  gemm_wmma_tdm    <<<gemm_blocks, blk, 0, stream>>>(xf16, wT16, h_buf, 128);

  alpha_kernel<<<cdiv(NH, TB), blk, 0, stream>>>(h_buf, a_src2, a_dst2, asb, adb);

  fill_kernel<<<cdiv(NH, TB), blk, 0, stream>>>(smax, -INFINITY, NH);
  fill_kernel<<<cdiv(NH, TB), blk, 0, stream>>>(den, 0.0f, NH);
  fill_kernel<<<cdiv(out_size, TB), blk, 0, stream>>>(out, 0.0f, out_size);

  edge_max_kernel  <<<cdiv((long long)E * HEADS, TB), blk, 0, stream>>>(ei, E_in, asb, adb, smax);
  clean_max_kernel <<<cdiv(NH, TB), blk, 0, stream>>>(smax);
  edge_denom_kernel<<<cdiv((long long)E * HEADS, TB), blk, 0, stream>>>(ei, E_in, asb, adb, smax, den);
  edge_agg_kernel  <<<cdiv(eagg, TB), blk, 0, stream>>>(ei, E_in, asb, adb, smax, den, h_buf, out);

  bias_elu_kernel<<<cdiv(ND, TB), blk, 0, stream>>>(out, b2, out, /*elu=*/0);
}
